// Attention_3539053052516
// MI455X (gfx1250) — compile-verified, matrix operand order checked by reading, not run on hardware
//
#include <hip/hip_runtime.h>
#include <hip/hip_bf16.h>
#include <cstddef>
#include <cstdint>

#define NHEAD 16
#define HD    64
#define NQ    2048
#define NKV   4096
#define CDIM  1024

typedef __attribute__((ext_vector_type(16))) __bf16 v16bf;
typedef __attribute__((ext_vector_type(8)))  float  v8f;

union Frag16 { v16bf v; uint4 q[2]; };

__device__ __forceinline__ unsigned short f2bf(float f) {
  union { float f; unsigned u; } x; x.f = f;
  unsigned r = x.u + 0x7FFFu + ((x.u >> 16) & 1u);
  return (unsigned short)(r >> 16);
}

// ---------------------------------------------------------------- convert
__global__ __launch_bounds__(256)
void cvt_f32_bf16(const float* __restrict__ src, unsigned short* __restrict__ dst, int n) {
  int i = blockIdx.x * blockDim.x + threadIdx.x;
  int stride = gridDim.x * blockDim.x;
  for (; i < n; i += stride) dst[i] = f2bf(src[i]);
}

// ---------------------------------------------------------------- GEMM: C[M][N] = A[M][K] * W[N][K]^T (+bias)
// 128 threads = 4 waves; block tile 128x64; wave tile 32x64 (2 row sub-tiles x 4 col tiles)
__global__ __launch_bounds__(128)
void gemm_bf16_wmma(const unsigned short* __restrict__ A,
                    const unsigned short* __restrict__ W,
                    float* __restrict__ C,
                    const float* __restrict__ bias,
                    int M, int N, int K) {
  const int tid  = threadIdx.x;
  const int wave = tid >> 5;
  const int lane = tid & 31;
  const int lh   = lane >> 4;   // lane-half
  const int m16  = lane & 15;
  const int row0 = blockIdx.y * 128 + wave * 32;
  const int col0 = blockIdx.x * 64;

  v8f acc[2][4] = {};
  const unsigned short* arow0 = A + (size_t)(row0 + m16) * K;
  const unsigned short* arow1 = A + (size_t)(row0 + 16 + m16) * K;

  for (int kc = 0; kc < K; kc += 32) {
    Frag16 a0, a1, b[4];
    a0.q[0] = *(const uint4*)(arow0 + kc + lh * 8);
    a0.q[1] = *(const uint4*)(arow0 + kc + 16 + lh * 8);
    a1.q[0] = *(const uint4*)(arow1 + kc + lh * 8);
    a1.q[1] = *(const uint4*)(arow1 + kc + 16 + lh * 8);
#pragma unroll
    for (int t = 0; t < 4; ++t) {
      const unsigned short* wr = W + (size_t)(col0 + 16 * t + m16) * K + kc;
      b[t].q[0] = *(const uint4*)(wr + lh * 8);
      b[t].q[1] = *(const uint4*)(wr + 16 + lh * 8);
    }
#pragma unroll
    for (int t = 0; t < 4; ++t) {
      acc[0][t] = __builtin_amdgcn_wmma_f32_16x16x32_bf16(
          false, a0.v, false, b[t].v, (short)0, acc[0][t], false, false);
      acc[1][t] = __builtin_amdgcn_wmma_f32_16x16x32_bf16(
          false, a1.v, false, b[t].v, (short)0, acc[1][t], false, false);
    }
  }
#pragma unroll
  for (int r = 0; r < 2; ++r)
#pragma unroll
    for (int t = 0; t < 4; ++t) {
      int col = col0 + 16 * t + m16;
      float bv = bias ? bias[col] : 0.0f;
#pragma unroll
      for (int j = 0; j < 8; ++j) {
        int row = row0 + 16 * r + j + 8 * lh;
        C[(size_t)row * N + col] = acc[r][t][j] + bv;
      }
    }
}

// ---------------------------------------------------------------- qkv(x) -> Q,Kx,Vx (rmsnorm + rotary, bf16 per-head)
__global__ __launch_bounds__(32)
void postproc_x(const float* __restrict__ qkv, const float* __restrict__ pos,
                const float* __restrict__ qw, const float* __restrict__ kw,
                unsigned short* __restrict__ Qb, unsigned short* __restrict__ Kb,
                unsigned short* __restrict__ Vb) {
  const int n = blockIdx.x >> 4;
  const int h = blockIdx.x & 15;
  const int i = threadIdx.x;          // lane handles d0=2i, d1=2i+1
  const int d0 = 2 * i, d1 = 2 * i + 1;
  const float* base = qkv + (size_t)n * 3072 + h * 64;
  const float c = pos[(size_t)n * 64 + 2 * i + 0];
  const float s = pos[(size_t)n * 64 + 2 * i + 1];
  { // q: norm + rotary
    float e = base[d0], o = base[d1];
    float ss = e * e + o * o;
    for (int msk = 1; msk < 32; msk <<= 1) ss += __shfl_xor(ss, msk, 32);
    float r = rsqrtf(ss * (1.0f / 64.0f) + 1e-6f);
    e = e * r * qw[d0]; o = o * r * qw[d1];
    size_t out = (size_t)h * NQ * HD + (size_t)n * HD;
    Qb[out + d0] = f2bf(e * c - o * s);
    Qb[out + d1] = f2bf(e * s + o * c);
  }
  { // kx: norm + rotary
    const float* kp = base + 1024;
    float e = kp[d0], o = kp[d1];
    float ss = e * e + o * o;
    for (int msk = 1; msk < 32; msk <<= 1) ss += __shfl_xor(ss, msk, 32);
    float r = rsqrtf(ss * (1.0f / 64.0f) + 1e-6f);
    e = e * r * kw[d0]; o = o * r * kw[d1];
    size_t out = (size_t)h * NKV * HD + (size_t)n * HD;
    Kb[out + d0] = f2bf(e * c - o * s);
    Kb[out + d1] = f2bf(e * s + o * c);
  }
  { // vx: passthrough
    const float* vp = base + 2048;
    size_t out = (size_t)h * NKV * HD + (size_t)n * HD;
    Vb[out + d0] = f2bf(vp[d0]);
    Vb[out + d1] = f2bf(vp[d1]);
  }
}

// ---------------------------------------------------------------- kv(y) -> Ky,Vy (norm only, appended at key offset 2048)
__global__ __launch_bounds__(32)
void postproc_y(const float* __restrict__ kv, const float* __restrict__ kw,
                unsigned short* __restrict__ Kb, unsigned short* __restrict__ Vb) {
  const int m = blockIdx.x >> 4;
  const int h = blockIdx.x & 15;
  const int i = threadIdx.x;
  const int d0 = 2 * i, d1 = 2 * i + 1;
  const float* base = kv + (size_t)m * 2048 + h * 64;
  {
    float e = base[d0], o = base[d1];
    float ss = e * e + o * o;
    for (int msk = 1; msk < 32; msk <<= 1) ss += __shfl_xor(ss, msk, 32);
    float r = rsqrtf(ss * (1.0f / 64.0f) + 1e-6f);
    size_t out = (size_t)h * NKV * HD + (size_t)(2048 + m) * HD;
    Kb[out + d0] = f2bf(e * r * kw[d0]);
    Kb[out + d1] = f2bf(o * r * kw[d1]);
  }
  {
    const float* vp = base + 1024;
    size_t out = (size_t)h * NKV * HD + (size_t)(2048 + m) * HD;
    Vb[out + d0] = f2bf(vp[d0]);
    Vb[out + d1] = f2bf(vp[d1]);
  }
}

// ---------------------------------------------------------------- flash attention (4 waves, 64 q-rows/block)
__global__ __launch_bounds__(128)
void flash_attn(const unsigned short* __restrict__ Q,
                const unsigned short* __restrict__ K,
                const unsigned short* __restrict__ V,
                unsigned short* __restrict__ O) {
  __shared__ unsigned short sK [64 * 64];   // [key][hd]
  __shared__ unsigned short sVt[64 * 64];   // [hd][key]
  __shared__ unsigned short sP [4 * 16 * 64];

  const int tid  = threadIdx.x;
  const int wave = tid >> 5;
  const int lane = tid & 31;
  const int lh   = lane >> 4;
  const int m16  = lane & 15;
  const int h    = blockIdx.y;
  const int q0   = blockIdx.x * 64 + wave * 16;
  const float scale = 0.125f;   // 1/sqrt(64)

  const unsigned short* Qh = Q + (size_t)h * NQ  * HD;
  const unsigned short* Kh = K + (size_t)h * NKV * HD;
  const unsigned short* Vh = V + (size_t)h * NKV * HD;

  Frag16 qa[2];                              // A-layout, hd chunks 0..31 / 32..63
  {
    const unsigned short* qrow = Qh + (size_t)(q0 + m16) * HD;
#pragma unroll
    for (int c2 = 0; c2 < 2; ++c2) {
      qa[c2].q[0] = *(const uint4*)(qrow + 32 * c2 + lh * 8);
      qa[c2].q[1] = *(const uint4*)(qrow + 32 * c2 + 16 + lh * 8);
    }
  }

  v8f o_acc[4] = {};
  float mrow[8], lrow[8];
#pragma unroll
  for (int j = 0; j < 8; ++j) { mrow[j] = -1e30f; lrow[j] = 0.0f; }

  unsigned short* pw = sP + wave * (16 * 64);

  for (int kt = 0; kt < NKV / 64; ++kt) {
    __syncthreads();
    { // stage K row-major, V transposed
      const unsigned short* kg = Kh + (size_t)kt * (64 * 64);
      const unsigned short* vg = Vh + (size_t)kt * (64 * 64);
      for (int i = tid; i < 512; i += 128) {
        *(uint4*)(sK + i * 8) = *(const uint4*)(kg + i * 8);
        uint4 vv = *(const uint4*)(vg + i * 8);
        const unsigned short* ve = (const unsigned short*)&vv;
        int key = i >> 3;
        int hb  = (i & 7) * 8;
#pragma unroll
        for (int e = 0; e < 8; ++e) sVt[(hb + e) * 64 + key] = ve[e];
      }
    }
    __syncthreads();

    if (kt + 1 < NKV / 64) {   // gfx1250 global_prefetch for next tile
      __builtin_prefetch(Kh + (size_t)(kt + 1) * (64 * 64) + tid * 32, 0, 1);
      __builtin_prefetch(Vh + (size_t)(kt + 1) * (64 * 64) + tid * 32, 0, 1);
    }

    // S = Q * K^T  (4 key sub-tiles of 16)
    v8f st[4];
#pragma unroll
    for (int t = 0; t < 4; ++t) {
      const unsigned short* kb = sK + (16 * t + m16) * 64;
      Frag16 b0, b1;
      b0.q[0] = *(const uint4*)(kb + lh * 8);
      b0.q[1] = *(const uint4*)(kb + 16 + lh * 8);
      b1.q[0] = *(const uint4*)(kb + 32 + lh * 8);
      b1.q[1] = *(const uint4*)(kb + 48 + lh * 8);
      v8f s = {};
      s = __builtin_amdgcn_wmma_f32_16x16x32_bf16(false, qa[0].v, false, b0.v, (short)0, s, false, false);
      s = __builtin_amdgcn_wmma_f32_16x16x32_bf16(false, qa[1].v, false, b1.v, (short)0, s, false, false);
      st[t] = s;
    }

    // online softmax (rows are per-VGPR; reduce across the 16-lane half)
#pragma unroll
    for (int j = 0; j < 8; ++j) {
      float mj = -1e30f;
#pragma unroll
      for (int t = 0; t < 4; ++t) mj = fmaxf(mj, st[t][j]);
      mj = fmaxf(mj, __shfl_xor(mj, 1, 32));
      mj = fmaxf(mj, __shfl_xor(mj, 2, 32));
      mj = fmaxf(mj, __shfl_xor(mj, 4, 32));
      mj = fmaxf(mj, __shfl_xor(mj, 8, 32));
      mj *= scale;
      float mnew  = fmaxf(mrow[j], mj);
      float alpha = __expf(mrow[j] - mnew);
      mrow[j] = mnew;
      float rs = 0.0f;
#pragma unroll
      for (int t = 0; t < 4; ++t) {
        float p = __expf(st[t][j] * scale - mnew);
        st[t][j] = p;
        rs += p;
      }
      rs += __shfl_xor(rs, 1, 32);
      rs += __shfl_xor(rs, 2, 32);
      rs += __shfl_xor(rs, 4, 32);
      rs += __shfl_xor(rs, 8, 32);
      lrow[j] = lrow[j] * alpha + rs;
#pragma unroll
      for (int u = 0; u < 4; ++u) o_acc[u][j] *= alpha;
    }

    // P: C-layout -> per-wave LDS -> A-layout
#pragma unroll
    for (int t = 0; t < 4; ++t)
#pragma unroll
      for (int j = 0; j < 8; ++j)
        pw[(j + 8 * lh) * 64 + 16 * t + m16] = f2bf(st[t][j]);
    asm volatile("s_wait_dscnt 0" ::: "memory");

    Frag16 pa[2];
    {
      const unsigned short* prow = pw + m16 * 64;
#pragma unroll
      for (int c2 = 0; c2 < 2; ++c2) {
        pa[c2].q[0] = *(const uint4*)(prow + 32 * c2 + lh * 8);
        pa[c2].q[1] = *(const uint4*)(prow + 32 * c2 + 16 + lh * 8);
      }
    }

    // O += P * V  (V already transposed in LDS -> contiguous B loads)
#pragma unroll
    for (int u = 0; u < 4; ++u) {
      const unsigned short* vb = sVt + (16 * u + m16) * 64;
      Frag16 b0, b1;
      b0.q[0] = *(const uint4*)(vb + lh * 8);
      b0.q[1] = *(const uint4*)(vb + 16 + lh * 8);
      b1.q[0] = *(const uint4*)(vb + 32 + lh * 8);
      b1.q[1] = *(const uint4*)(vb + 48 + lh * 8);
      o_acc[u] = __builtin_amdgcn_wmma_f32_16x16x32_bf16(false, pa[0].v, false, b0.v, (short)0, o_acc[u], false, false);
      o_acc[u] = __builtin_amdgcn_wmma_f32_16x16x32_bf16(false, pa[1].v, false, b1.v, (short)0, o_acc[u], false, false);
    }
  }

  // epilogue: O/l -> bf16, interleave heads back into [N][C]
#pragma unroll
  for (int u = 0; u < 4; ++u)
#pragma unroll
    for (int j = 0; j < 8; ++j) {
      int row = q0 + j + 8 * lh;
      float inv = 1.0f / lrow[j];
      O[(size_t)row * CDIM + h * 64 + 16 * u + m16] = f2bf(o_acc[u][j] * inv);
    }
}

// ---------------------------------------------------------------- host
extern "C" void kernel_launch(void* const* d_in, const int* in_sizes, int n_in,
                              void* d_out, int out_size, void* d_ws, size_t ws_size,
                              hipStream_t stream) {
  const float* x     = (const float*)d_in[0];
  const float* y     = (const float*)d_in[1];
  const float* pos   = (const float*)d_in[2];
  const float* wqkv  = (const float*)d_in[3];
  const float* wkv   = (const float*)d_in[4];
  const float* wproj = (const float*)d_in[5];
  const float* bproj = (const float*)d_in[6];
  const float* qw    = (const float*)d_in[7];
  const float* kw    = (const float*)d_in[8];
  float* out = (float*)d_out;

  char* ws = (char*)d_ws;
  size_t off = 0;
  auto alloc = [&](size_t bytes) {
    char* p = ws + off;
    off += (bytes + 255) & ~(size_t)255;
    return p;
  };
  unsigned short* xb    = (unsigned short*)alloc((size_t)NQ * CDIM * 2);
  unsigned short* yb    = (unsigned short*)alloc((size_t)NQ * CDIM * 2);
  unsigned short* wqkvb = (unsigned short*)alloc((size_t)3 * CDIM * CDIM * 2);
  unsigned short* wkvb  = (unsigned short*)alloc((size_t)2 * CDIM * CDIM * 2);
  unsigned short* wprjb = (unsigned short*)alloc((size_t)CDIM * CDIM * 2);
  float*          qkvf  = (float*)alloc((size_t)NQ * 3 * CDIM * 4);
  float*          kvf   = (float*)alloc((size_t)NQ * 2 * CDIM * 4);
  unsigned short* Qb    = (unsigned short*)alloc((size_t)NHEAD * NQ  * HD * 2);
  unsigned short* Kb    = (unsigned short*)alloc((size_t)NHEAD * NKV * HD * 2);
  unsigned short* Vb    = (unsigned short*)alloc((size_t)NHEAD * NKV * HD * 2);
  unsigned short* Ob    = (unsigned short*)alloc((size_t)NQ * CDIM * 2);

  cvt_f32_bf16<<<1024, 256, 0, stream>>>(x,     xb,    NQ * CDIM);
  cvt_f32_bf16<<<1024, 256, 0, stream>>>(y,     yb,    NQ * CDIM);
  cvt_f32_bf16<<<1024, 256, 0, stream>>>(wqkv,  wqkvb, 3 * CDIM * CDIM);
  cvt_f32_bf16<<<1024, 256, 0, stream>>>(wkv,   wkvb,  2 * CDIM * CDIM);
  cvt_f32_bf16<<<1024, 256, 0, stream>>>(wproj, wprjb, CDIM * CDIM);

  gemm_bf16_wmma<<<dim3(48, 16), 128, 0, stream>>>(xb, wqkvb, qkvf, nullptr,
                                                   NQ, 3 * CDIM, CDIM);
  gemm_bf16_wmma<<<dim3(32, 16), 128, 0, stream>>>(yb, wkvb, kvf, nullptr,
                                                   NQ, 2 * CDIM, CDIM);

  postproc_x<<<NQ * NHEAD, 32, 0, stream>>>(qkvf, pos, qw, kw, Qb, Kb, Vb);
  postproc_y<<<NQ * NHEAD, 32, 0, stream>>>(kvf, kw, Kb, Vb);

  flash_attn<<<dim3(NQ / 64, NHEAD), 128, 0, stream>>>(Qb, Kb, Vb, Ob);

  gemm_bf16_wmma<<<dim3(16, 16), 128, 0, stream>>>(Ob, wprjb, out, bproj,
                                                   NQ, CDIM, CDIM);
}
